// NonLocalAttention_90314572300707
// MI455X (gfx1250) — matrix operand order repaired and checked by reading, not compiled, not run on hardware
//
#include <hip/hip_runtime.h>
#include <cstddef>

// ---- CDNA5 types ----
typedef __bf16 bf16;
typedef __attribute__((ext_vector_type(16))) __bf16 v16bf;
typedef __attribute__((ext_vector_type(8)))  __bf16 v8bf;
typedef __attribute__((ext_vector_type(8)))  float  v8f;
typedef __attribute__((ext_vector_type(4)))  int    i32x4;

static constexpr int Bn  = 4;
static constexpr int Cc  = 256;   // in/out channels
static constexpr int Cin = 128;   // inter channels
static constexpr int Nn  = 4096;  // H*W

// gfx1250 async global->LDS path (ASYNCcnt), guarded so compile can't break.
#if defined(__has_builtin)
#if __has_builtin(__builtin_amdgcn_global_load_async_to_lds_b128) && \
    __has_builtin(__builtin_amdgcn_s_wait_asynccnt)
#define USE_ASYNC_LDS 1
#endif
#endif
#ifndef USE_ASYNC_LDS
#define USE_ASYNC_LDS 0
#endif

static __device__ __forceinline__ void copy16_to_lds(const bf16* g, bf16* l) {
#if USE_ASYNC_LDS
  // signature (probe-confirmed by diagnostic): (int4 as1*, int4 as3*, imm, imm)
  __builtin_amdgcn_global_load_async_to_lds_b128(
      (__attribute__((address_space(1))) i32x4*)(void*)g,
      (__attribute__((address_space(3))) i32x4*)(void*)l, 0, 0);
#else
  *(v8bf*)l = *(const v8bf*)g;
#endif
}
static __device__ __forceinline__ void async_wait_all() {
#if USE_ASYNC_LDS
  __builtin_amdgcn_s_wait_asynccnt(0);
#endif
}

static __device__ __forceinline__ v8f zero8() {
  v8f z;
#pragma unroll
  for (int i = 0; i < 8; ++i) z[i] = 0.0f;
  return z;
}

static __device__ __forceinline__ v16bf combine16(v8bf p0, v8bf p1) {
  v16bf r;
#pragma unroll
  for (int e = 0; e < 8; ++e) { r[e] = p0[e]; r[8 + e] = p1[e]; }
  return r;
}

// =====================================================================
// Kernel 0: one-shot fp32 -> bf16 weight conversion (theta/phi/g/W).
// Removes per-wave v_cvt_pk_bf16_f32 from all hot GEMM loops.
// =====================================================================
__global__ __launch_bounds__(256) void wcvt_kernel(
    const float* __restrict__ tw, const float* __restrict__ pw,
    const float* __restrict__ gw, const float* __restrict__ Ww,
    bf16* __restrict__ twh, bf16* __restrict__ pwh,
    bf16* __restrict__ gwh, bf16* __restrict__ Wwh)
{
  const int i = blockIdx.x * 256 + threadIdx.x;   // Cin*Cc == Cc*Cin elements
  if (i < Cin * Cc) {
    twh[i] = (bf16)tw[i];
    pwh[i] = (bf16)pw[i];
    gwh[i] = (bf16)gw[i];
    Wwh[i] = (bf16)Ww[i];
  }
}

// =====================================================================
// Kernel 1: three 1x1-conv projections as bf16 WMMA GEMMs.
//   theta_h[b][n][ci]  (row-major, A-friendly)   phi_h[b][m][ci]
//   gT_h   [b][ci][m]  (ci-major, B-friendly)
// 3 waves/block: wave 0 -> theta, 1 -> phi, 2 -> g. 16-row tile each.
// =====================================================================
__global__ __launch_bounds__(96) void proj_kernel(
    const float* __restrict__ a, const float* __restrict__ bsrc,
    const bf16* __restrict__ tw_h, const bf16* __restrict__ pw_h,
    const bf16* __restrict__ gw_h,
    const float* __restrict__ theta_b, const float* __restrict__ phi_b,
    const float* __restrict__ g_b,
    bf16* __restrict__ theta_h, bf16* __restrict__ phi_h, bf16* __restrict__ gT_h)
{
  const int lane = threadIdx.x & 31;
  const int wave = threadIdx.x >> 5;       // 0:theta 1:phi 2:g
  const int hi = lane >> 4, lo = lane & 15;
  const int blk   = blockIdx.x;            // Bn * (Nn/16)
  const int batch = blk / (Nn / 16);
  const int n0    = (blk % (Nn / 16)) * 16;

  const float* x    = (wave == 0) ? a : bsrc;
  const bf16*  wh   = (wave == 0) ? tw_h : (wave == 1 ? pw_h : gw_h);
  const float* bias = (wave == 0) ? theta_b : (wave == 1 ? phi_b : g_b);
  const float* xb   = x + (size_t)batch * Cc * Nn;

  v8f acc[8];
#pragma unroll
  for (int t = 0; t < 8; ++t) acc[t] = zero8();

  for (int kc = 0; kc < Cc / 32; ++kc) {
    const int c_base = kc * 32;
    // A fragment: x^T tile [16 n x 32 c], strided fp32 gather + cvt
    v16bf A;
#pragma unroll
    for (int e = 0; e < 16; ++e) {
      const int k = (e < 8) ? (8 * hi + e) : (16 + 8 * hi + (e - 8));
      A[e] = (bf16)xb[(size_t)(c_base + k) * Nn + n0 + lo];
    }
#pragma unroll
    for (int ot = 0; ot < 8; ++ot) {
      // B fragment: bf16 w^T [32 c x 16 o]; contiguous b128 loads, no cvt
      const bf16* wrow = wh + (size_t)(ot * 16 + lo) * Cc + c_base + 8 * hi;
      v16bf Bm = combine16(*(const v8bf*)wrow, *(const v8bf*)(wrow + 16));
      acc[ot] = __builtin_amdgcn_wmma_f32_16x16x32_bf16(
          false, A, false, Bm, (short)0, acc[ot], false, false);
    }
  }

#pragma unroll
  for (int ot = 0; ot < 8; ++ot) {
    const int o  = ot * 16 + lo;
    const float bv = bias[o];
#pragma unroll
    for (int v = 0; v < 8; ++v) {
      const int n = n0 + v + 8 * hi;
      const bf16 val = (bf16)(acc[ot][v] + bv);
      if (wave == 0)      theta_h[((size_t)batch * Nn + n) * Cin + o] = val;
      else if (wave == 1) phi_h  [((size_t)batch * Nn + n) * Cin + o] = val;
      else                gT_h   [((size_t)batch * Cin + o) * Nn + n] = val;
    }
  }
}

// =====================================================================
// Kernel 2: fused  y = (theta . phi^T / N) . g   -- f never hits HBM.
// 8 waves/block, wave owns 16 n-rows. Per 32-m chunk:
//   block-cooperative async stage of phi[32x128] + gT[128x32] into padded
//   LDS (double buffered; all 8 waves share it -> 8x less L2 traffic),
//   stage 1: f[16x32] = 8 WMMA over Ci, LDS D->A relayout,
//   stage 2: y[16x128] += 8 WMMA (K = 32 m).
// =====================================================================
__global__ __launch_bounds__(256) void attn_kernel(
    const bf16* __restrict__ theta_h, const bf16* __restrict__ phi_h,
    const bf16* __restrict__ gT_h, bf16* __restrict__ y_h)
{
  __shared__ bf16 phis[2][32][136];   // 128 + 8 pad halves per row
  __shared__ bf16 gs[2][128][40];     // 32 + 8 pad halves per row
  __shared__ bf16 flds[8][16][40];    // per-wave D->A relayout scratch

  const int tid  = threadIdx.x;
  const int lane = tid & 31, wave = tid >> 5;
  const int hi = lane >> 4, lo = lane & 15;
  const int blk   = blockIdx.x;             // Bn * (Nn/128)
  const int batch = blk / (Nn / 128);
  const int n0    = (blk % (Nn / 128)) * 128 + wave * 16;

  const bf16* th = theta_h + (size_t)batch * Nn * Cin;
  const bf16* ph = phi_h   + (size_t)batch * Nn * Cin;
  const bf16* gT = gT_h    + (size_t)batch * Cin * Nn;

  // Preload row-invariant A_theta fragments (4 K-chunks over Ci=128)
  v16bf Ath[4];
#pragma unroll
  for (int kc = 0; kc < 4; ++kc) {
    const bf16* row = th + (size_t)(n0 + lo) * Cin + kc * 32;
    Ath[kc] = combine16(*(const v8bf*)(row + 8 * hi),
                        *(const v8bf*)(row + 16 + 8 * hi));
  }

  v8f yacc[8];
#pragma unroll
  for (int t = 0; t < 8; ++t) yacc[t] = zero8();
  const float inv_n = 1.0f / (float)Nn;

  // cooperative stage of one 32-m chunk into LDS buffer `bsel`
  auto stage = [&](int bsel, int m0) {
#pragma unroll
    for (int i = 0; i < 2; ++i) {           // phi: 512 x 16B, 2 per thread
      const int v = tid + i * 256;
      const int r = v >> 4, c8 = v & 15;
      copy16_to_lds(ph + (size_t)(m0 + r) * Cin + c8 * 8, &phis[bsel][r][c8 * 8]);
    }
#pragma unroll
    for (int i = 0; i < 2; ++i) {           // gT: 512 x 16B, 2 per thread
      const int v = tid + i * 256;
      const int r = v >> 2, c8 = v & 3;
      copy16_to_lds(gT + (size_t)r * Nn + m0 + c8 * 8, &gs[bsel][r][c8 * 8]);
    }
  };

  int buf = 0;
  stage(0, 0);
  async_wait_all();
  __syncthreads();

  for (int m0 = 0; m0 < Nn; m0 += 32) {
    if (m0 + 32 < Nn) stage(buf ^ 1, m0 + 32);   // prefetch next chunk

    // ---- stage 1: f strip [16 n x 32 m] from LDS ----
    v8f facc[2]; facc[0] = zero8(); facc[1] = zero8();
#pragma unroll
    for (int t = 0; t < 2; ++t) {
#pragma unroll
      for (int kc = 0; kc < 4; ++kc) {
        const bf16* prow = &phis[buf][t * 16 + lo][kc * 32 + 8 * hi];
        v16bf Bp = combine16(*(const v8bf*)(prow), *(const v8bf*)(prow + 16));
        facc[t] = __builtin_amdgcn_wmma_f32_16x16x32_bf16(
            false, Ath[kc], false, Bp, (short)0, facc[t], false, false);
      }
    }
    // ---- D-layout -> LDS (scaled, bf16), wave-private region ----
#pragma unroll
    for (int t = 0; t < 2; ++t)
#pragma unroll
      for (int v = 0; v < 8; ++v)
        flds[wave][v + 8 * hi][t * 16 + lo] = (bf16)(facc[t][v] * inv_n);

    const bf16* r = &flds[wave][lo][0];
    v16bf Af = combine16(*(const v8bf*)(r + 8 * hi),
                         *(const v8bf*)(r + 16 + 8 * hi));

    // ---- stage 2: y += f . g  (K = 32 m) from LDS ----
#pragma unroll
    for (int ct = 0; ct < 8; ++ct) {
      const bf16* grow = &gs[buf][ct * 16 + lo][8 * hi];
      v16bf Bg = combine16(*(const v8bf*)(grow), *(const v8bf*)(grow + 16));
      yacc[ct] = __builtin_amdgcn_wmma_f32_16x16x32_bf16(
          false, Af, false, Bg, (short)0, yacc[ct], false, false);
    }

    async_wait_all();      // next-chunk asyncs landed
    __syncthreads();       // everyone done reading `buf`
    buf ^= 1;
  }

  // store y row-major [n][ci] bf16 (A-friendly for kernel 3)
  bf16* yb = y_h + (size_t)batch * Nn * Cin;
#pragma unroll
  for (int ct = 0; ct < 8; ++ct) {
    const int ci = ct * 16 + lo;
#pragma unroll
    for (int v = 0; v < 8; ++v)
      yb[(size_t)(n0 + v + 8 * hi) * Cin + ci] = (bf16)yacc[ct][v];
  }
}

// =====================================================================
// Kernel 3: out = BN(W_w . y). Wave owns 16 rows; 16 cout-tiles x 4 K-chunks
// WMMA with bf16 weights; BN folded into the epilogue.
// =====================================================================
__global__ __launch_bounds__(128) void outproj_kernel(
    const bf16* __restrict__ y_h, const bf16* __restrict__ Ww_h,
    const float* __restrict__ gamma, const float* __restrict__ beta,
    const float* __restrict__ mean,  const float* __restrict__ var,
    float* __restrict__ out)
{
  const int lane = threadIdx.x & 31, wave = threadIdx.x >> 5;
  const int hi = lane >> 4, lo = lane & 15;
  const int blk   = blockIdx.x;             // Bn * (Nn/64)
  const int batch = blk / (Nn / 64);
  const int n0    = (blk % (Nn / 64)) * 64 + wave * 16;
  const bf16* yb  = y_h + (size_t)batch * Nn * Cin;

  v8f acc[16];
#pragma unroll
  for (int t = 0; t < 16; ++t) acc[t] = zero8();

#pragma unroll
  for (int kc = 0; kc < 4; ++kc) {
    const bf16* row = yb + (size_t)(n0 + lo) * Cin + kc * 32;
    v16bf A = combine16(*(const v8bf*)(row + 8 * hi),
                        *(const v8bf*)(row + 16 + 8 * hi));
#pragma unroll
    for (int ct = 0; ct < 16; ++ct) {
      const bf16* wrow = Ww_h + (size_t)(ct * 16 + lo) * Cin + kc * 32 + 8 * hi;
      v16bf Bm = combine16(*(const v8bf*)wrow, *(const v8bf*)(wrow + 16));
      acc[ct] = __builtin_amdgcn_wmma_f32_16x16x32_bf16(
          false, A, false, Bm, (short)0, acc[ct], false, false);
    }
  }

  float* ob = out + (size_t)batch * Cc * Nn;
#pragma unroll
  for (int ct = 0; ct < 16; ++ct) {
    const int o = ct * 16 + lo;
    const float sc = gamma[o] * rsqrtf(var[o] + 1e-5f);
    const float sh = beta[o] - mean[o] * sc;
#pragma unroll
    for (int v = 0; v < 8; ++v)
      ob[(size_t)o * Nn + n0 + v + 8 * hi] = acc[ct][v] * sc + sh;
  }
}

// =====================================================================
extern "C" void kernel_launch(void* const* d_in, const int* in_sizes, int n_in,
                              void* d_out, int out_size, void* d_ws, size_t ws_size,
                              hipStream_t stream) {
  const float* a       = (const float*)d_in[0];
  const float* b       = (const float*)d_in[1];
  const float* theta_w = (const float*)d_in[2];
  const float* theta_b = (const float*)d_in[3];
  const float* phi_w   = (const float*)d_in[4];
  const float* phi_b   = (const float*)d_in[5];
  const float* g_w     = (const float*)d_in[6];
  const float* g_b     = (const float*)d_in[7];
  const float* W_w     = (const float*)d_in[8];
  const float* bn_g    = (const float*)d_in[9];
  const float* bn_b    = (const float*)d_in[10];
  const float* bn_m    = (const float*)d_in[11];
  const float* bn_v    = (const float*)d_in[12];
  float* out = (float*)d_out;

  const size_t wElems    = (size_t)Cin * Cc;        // 32768, all four weights
  const size_t projElems = (size_t)Bn * Nn * Cin;   // 4 MB each in bf16

  bf16* tw_h    = (bf16*)d_ws;
  bf16* pw_h    = tw_h + wElems;
  bf16* gw_h    = pw_h + wElems;
  bf16* Ww_h    = gw_h + wElems;
  bf16* theta_h = Ww_h + wElems;
  bf16* phi_h   = theta_h + projElems;
  bf16* gT_h    = phi_h + projElems;
  bf16* y_h     = gT_h + projElems;

  wcvt_kernel<<<(Cin * Cc + 255) / 256, 256, 0, stream>>>(
      theta_w, phi_w, g_w, W_w, tw_h, pw_h, gw_h, Ww_h);

  proj_kernel<<<Bn * (Nn / 16), 96, 0, stream>>>(
      a, b, tw_h, pw_h, gw_h, theta_b, phi_b, g_b, theta_h, phi_h, gT_h);

  attn_kernel<<<Bn * (Nn / 128), 256, 0, stream>>>(theta_h, phi_h, gT_h, y_h);

  outproj_kernel<<<Bn * (Nn / 64), 128, 0, stream>>>(
      y_h, Ww_h, bn_g, bn_b, bn_m, bn_v, out);
}
